// SambaYPureAttention_40097814675978
// MI455X (gfx1250) — matrix operand order repaired and checked by reading, not compiled, not run on hardware
//
#include <hip/hip_runtime.h>
#include <hip/hip_bf16.h>
#include <math.h>

typedef __attribute__((ext_vector_type(16))) _Float16     v16h;
typedef __attribute__((ext_vector_type(8)))  float        v8f;
typedef __attribute__((ext_vector_type(4)))  unsigned int v4u;
typedef __attribute__((ext_vector_type(8)))  int          v8i;
typedef __attribute__((ext_vector_type(4)))  int          v4i;

#define NUM_HEADS   32
#define NUM_KV      8
#define HEAD_DIM    64
#define SEG_LEN     1024
#define TOKENS      4096
#define HDIM        2048
#define KVDIM       512

static __device__ inline v8f v8f_zero() {
    v8f z = {0.f, 0.f, 0.f, 0.f, 0.f, 0.f, 0.f, 0.f};
    return z;
}

// ---------------------------------------------------------------------------
// Tensor Data Mover: DMA a 2-D tile (tile_d1 rows x tile_d0 elements, f16)
// from global memory into LDS at byte offset lds_off. D# per CDNA5 ISA ch.8.
// ---------------------------------------------------------------------------
static __device__ inline void tdm_load_2d_f16(unsigned lds_off, const void* gptr,
                                              unsigned tensor_d0, unsigned tensor_d1,
                                              unsigned tile_d0, unsigned tile_d1,
                                              unsigned stride_d0) {
    unsigned long long ga = (unsigned long long)gptr;
    v4u g0;
    g0.x = 1u;                                   // count=1, user descriptor
    g0.y = lds_off;                              // lds_addr (bytes)
    g0.z = (unsigned)(ga & 0xFFFFFFFFu);         // global_addr[31:0]
    g0.w = (unsigned)((ga >> 32) & 0x01FFFFFFu)  // global_addr[56:32]
         | (2u << 30);                           // type = 2 ("image")
    v8i g1;
    g1[0] = (int)(1u << 16);                     // wg_mask=0, data_size=1 (2 bytes)
    g1[1] = (int)((tensor_d0 & 0xFFFFu) << 16);  // tensor_dim0[15:0] in [31:16]
    g1[2] = (int)(((tensor_d0 >> 16) & 0xFFFFu)
         | ((tensor_d1 & 0xFFFFu) << 16));       // tensor_dim0[31:16] | tensor_dim1[15:0]
    g1[3] = (int)(((tensor_d1 >> 16) & 0xFFFFu)
         | ((tile_d0 & 0xFFFFu) << 16));         // tensor_dim1[31:16] | tile_dim0
    g1[4] = (int)(tile_d1 & 0xFFFFu);            // tile_dim1 | tile_dim2=0
    g1[5] = (int)stride_d0;                      // tensor_dim0_stride[31:0]
    g1[6] = 0;                                   // stride0[47:32] | dim1_stride[15:0]
    g1[7] = 0;                                   // dim1_stride[47:16]
    v4i z4 = {0, 0, 0, 0};
#if __clang_major__ >= 23
    v8i z8 = {0, 0, 0, 0, 0, 0, 0, 0};
    __builtin_amdgcn_tensor_load_to_lds(g0, g1, z4, z4, z8, 0);
#else
    __builtin_amdgcn_tensor_load_to_lds(g0, g1, z4, z4, 0);
#endif
}

// ---------------------------------------------------------------------------
// WMMA fragment loaders (wave32; contiguous k order, same in A and B)
// A (16x32 f16): lane l -> row m=l&15, g=l>>4; element e -> K = g*16 + e
// ---------------------------------------------------------------------------
static __device__ inline v16h load_a_frag(const _Float16* A, int lda, int lane) {
    const int m = lane & 15, g = lane >> 4;
    const _Float16* p = A + (size_t)m * lda;
    v16h a;
#pragma unroll
    for (int e = 0; e < 8; ++e) a[e] = p[g * 16 + e];
#pragma unroll
    for (int e = 0; e < 8; ++e) a[8 + e] = p[g * 16 + 8 + e];
    return a;
}

// B (32x16 f16, KxN), logical B = W^T with W (N,K) row-major:
// lane l -> col n=l&15, g=l>>4; element e -> K = g*16 + e (16 contiguous f16).
static __device__ inline v16h load_b_frag(const _Float16* W, int ldw, int lane) {
    const int n = lane & 15, g = lane >> 4;
    const _Float16* p = W + (size_t)n * ldw + g * 16;
    v16h b;
#pragma unroll
    for (int e = 0; e < 16; ++e) b[e] = p[e];
    return b;
}

// ---------------------------------------------------------------------------
__global__ void f32_to_f16_kernel(const float* __restrict__ in, _Float16* __restrict__ out, int n) {
    int i = blockIdx.x * blockDim.x + threadIdx.x;
    if (i < n) out[i] = (_Float16)in[i];
}

// ---------------------------------------------------------------------------
// C(MxN) = A(MxK) * W(NxK)^T   -- f16 in, f32 accumulate, OutT out.
// 256 threads = 8 waves; block tile 128x128; wave tile 32x64.
// Operands live in L2 (192MB); direct clause'd b128 loads + prefetch.
// ---------------------------------------------------------------------------
template <typename OutT>
__global__ void __launch_bounds__(256)
wmma_gemm_nt(const _Float16* __restrict__ A, const _Float16* __restrict__ W,
             OutT* __restrict__ C, int M, int N, int K) {
    const int lane = threadIdx.x & 31;
    const int wave = threadIdx.x >> 5;
    const int row0 = blockIdx.y * 128 + (wave >> 1) * 32;
    const int col0 = blockIdx.x * 128 + (wave & 1) * 64;

    v8f acc[2][4];
#pragma unroll
    for (int i = 0; i < 2; ++i)
#pragma unroll
        for (int j = 0; j < 4; ++j) acc[i][j] = v8f_zero();

    const _Float16* Arow = A + (size_t)row0 * K;
    const _Float16* Wrow = W + (size_t)col0 * K;

    for (int k0 = 0; k0 < K; k0 += 32) {
        __builtin_prefetch(Arow + k0 + 64, 0, 0);
        __builtin_prefetch(Wrow + k0 + 64, 0, 0);
        v16h a0 = load_a_frag(Arow + k0, K, lane);
        v16h a1 = load_a_frag(Arow + (size_t)16 * K + k0, K, lane);
#pragma unroll
        for (int j = 0; j < 4; ++j) {
            v16h b = load_b_frag(Wrow + (size_t)(j * 16) * K + k0, K, lane);
            acc[0][j] = __builtin_amdgcn_wmma_f32_16x16x32_f16(
                false, a0, false, b, (short)0, acc[0][j], false, false);
            acc[1][j] = __builtin_amdgcn_wmma_f32_16x16x32_f16(
                false, a1, false, b, (short)0, acc[1][j], false, false);
        }
    }

    const int g = lane >> 4, n = lane & 15;
#pragma unroll
    for (int i = 0; i < 2; ++i)
#pragma unroll
        for (int j = 0; j < 4; ++j)
#pragma unroll
            for (int r = 0; r < 8; ++r)
                C[(size_t)(row0 + i * 16 + g * 8 + r) * N + col0 + j * 16 + n] =
                    (OutT)acc[i][j][r];
}

// ---------------------------------------------------------------------------
// Flash attention, one wave per (segment, 16-row q-tile, head).
// K/V 32x64 tiles are double-buffered in LDS and streamed by the Tensor Data
// Mover: while computing on buffer b, the TDM fills buffer b^1; in-order
// TENSORcnt retirement means s_wait_tensorcnt 2 = "current buffer ready".
// Buffer selection is done with integer offsets (GEP on the shared array) so
// address-space inference keeps all fragment reads on the DS path.
// ---------------------------------------------------------------------------
#define KVBUF 2048                       // one 32x64 f16 tile, in halves
#define WSZ   (4 * KVBUF + 512)          // K0,K1,V0,V1 + P staging (halves)

__global__ void __launch_bounds__(64)
attn_kernel(const _Float16* __restrict__ Q, const _Float16* __restrict__ Km,
            const _Float16* __restrict__ Vm, _Float16* __restrict__ O) {
    __shared__ __align__(16) _Float16 smem[2 * WSZ];   // 2 waves per block

    const int lane = threadIdx.x & 31;
    const int wave = threadIdx.x >> 5;
    _Float16* base = smem + wave * WSZ;                 // K0 | K1 | V0 | V1 | P
    _Float16* pbuf = base + 4 * KVBUF;
    const unsigned base_off = (unsigned)(wave * WSZ * 2);   // bytes

    const int gid  = blockIdx.x * 2 + wave;
    const int head = gid & (NUM_HEADS - 1);
    const int t    = gid / NUM_HEADS;
    const int qt   = t & 63;
    const int seg  = t >> 6;
    const int segbase = seg * SEG_LEN;
    const int q0   = segbase + qt * 16;
    const int kvh  = head >> 2;
    const int g = lane >> 4, n = lane & 15;

    const _Float16* Kg0 = Km + (size_t)segbase * KVDIM + kvh * HEAD_DIM;
    const _Float16* Vg0 = Vm + (size_t)segbase * KVDIM + kvh * HEAD_DIM;
    const int nkb = ((qt + 1) * 16 + 31) >> 5;  // 32-token k blocks

    // ---- prologue: start DMA of k-block 0 before touching Q ----
    tdm_load_2d_f16(base_off,                 Kg0, KVDIM, TOKENS, HEAD_DIM, 32, KVDIM);
    tdm_load_2d_f16(base_off + 4u * KVBUF,    Vg0, KVDIM, TOKENS, HEAD_DIM, 32, KVDIM);

    // Q fragments, pre-scaled by 1/sqrt(64)
    const _Float16* Qp = Q + (size_t)q0 * HDIM + head * HEAD_DIM;
    v16h aq0 = load_a_frag(Qp,      HDIM, lane);
    v16h aq1 = load_a_frag(Qp + 32, HDIM, lane);
#pragma unroll
    for (int e = 0; e < 16; ++e) {
        aq0[e] = (_Float16)((float)aq0[e] * 0.125f);
        aq1[e] = (_Float16)((float)aq1[e] * 0.125f);
    }

    v8f acc[4];
#pragma unroll
    for (int j = 0; j < 4; ++j) acc[j] = v8f_zero();
    float mrow[8], lrow[8], alpha[8];
#pragma unroll
    for (int r = 0; r < 8; ++r) { mrow[r] = -1e30f; lrow[r] = 0.f; }

    for (int kb = 0; kb < nkb; ++kb) {
        const int kc0 = kb * 32;
        const int cur = kb & 1;
        const int curK = cur * KVBUF;              // halves, offset of K buffer
        const int curV = 2 * KVBUF + cur * KVBUF;  // halves, offset of V buffer

        // ---- issue next tile's DMA, then wait only for the current tile ----
        if (kb + 1 < nkb) {
            const int nxt = cur ^ 1;
            const _Float16* Kg = Kg0 + (size_t)(kc0 + 32) * KVDIM;
            const _Float16* Vg = Vg0 + (size_t)(kc0 + 32) * KVDIM;
            tdm_load_2d_f16(base_off + (unsigned)nxt * KVBUF * 2u,
                            Kg, KVDIM, TOKENS, HEAD_DIM, 32, KVDIM);
            tdm_load_2d_f16(base_off + (unsigned)(2 + nxt) * KVBUF * 2u,
                            Vg, KVDIM, TOKENS, HEAD_DIM, 32, KVDIM);
            __builtin_amdgcn_s_wait_tensorcnt(2);   // current buffer complete
        } else {
            __builtin_amdgcn_s_wait_tensorcnt(0);
        }

        // ---- scores: two 16-col tiles, depth 64 = 2 chained WMMAs each ----
        v8f s[2];
#pragma unroll
        for (int jj = 0; jj < 2; ++jj) {
            v16h b0 = load_b_frag(base + curK + jj * 16 * HEAD_DIM,      HEAD_DIM, lane);
            v16h b1 = load_b_frag(base + curK + jj * 16 * HEAD_DIM + 32, HEAD_DIM, lane);
            v8f c = v8f_zero();
            c = __builtin_amdgcn_wmma_f32_16x16x32_f16(false, aq0, false, b0,
                                                       (short)0, c, false, false);
            c = __builtin_amdgcn_wmma_f32_16x16x32_f16(false, aq1, false, b1,
                                                       (short)0, c, false, false);
            s[jj] = c;
        }

        // ---- online softmax (rows span 16 lanes of a half-wave) ----
#pragma unroll
        for (int r = 0; r < 8; ++r) {
            const int qpos = qt * 16 + g * 8 + r;
            float s0 = s[0][r];
            float s1 = s[1][r];
            if (kc0 + n      > qpos) s0 = -1e30f;   // causal mask
            if (kc0 + 16 + n > qpos) s1 = -1e30f;

            float mx = fmaxf(s0, s1);
            mx = fmaxf(mx, __shfl_xor(mx, 1));
            mx = fmaxf(mx, __shfl_xor(mx, 2));
            mx = fmaxf(mx, __shfl_xor(mx, 4));
            mx = fmaxf(mx, __shfl_xor(mx, 8));
            const float mnew = fmaxf(mrow[r], mx);

            const float p0 = expf(s0 - mnew);
            const float p1 = expf(s1 - mnew);
            float rs = p0 + p1;
            rs += __shfl_xor(rs, 1);
            rs += __shfl_xor(rs, 2);
            rs += __shfl_xor(rs, 4);
            rs += __shfl_xor(rs, 8);

            alpha[r] = expf(mrow[r] - mnew);
            lrow[r]  = lrow[r] * alpha[r] + rs;
            mrow[r]  = mnew;

            pbuf[(g * 8 + r) * 32 + n]      = (_Float16)p0;
            pbuf[(g * 8 + r) * 32 + 16 + n] = (_Float16)p1;
        }

        asm volatile("s_wait_dscnt 0" ::: "memory");
        const v16h ap = load_a_frag(pbuf, 32, lane);   // P as A-fragment (16x32)

        // ---- O = O*alpha + P * V (V B-fragments gathered from LDS) ----
#pragma unroll
        for (int vt = 0; vt < 4; ++vt) {
#pragma unroll
            for (int r = 0; r < 8; ++r) acc[vt][r] *= alpha[r];
            v16h bv;
#pragma unroll
            for (int e = 0; e < 16; ++e)
                bv[e] = base[curV + (g * 16 + e) * HEAD_DIM + vt * 16 + n];
            acc[vt] = __builtin_amdgcn_wmma_f32_16x16x32_f16(
                false, ap, false, bv, (short)0, acc[vt], false, false);
        }
    }

#pragma unroll
    for (int vt = 0; vt < 4; ++vt)
#pragma unroll
        for (int r = 0; r < 8; ++r)
            O[(size_t)(q0 + g * 8 + r) * HDIM + head * HEAD_DIM + vt * 16 + n] =
                (_Float16)(acc[vt][r] / lrow[r]);
}

// ---------------------------------------------------------------------------
extern "C" void kernel_launch(void* const* d_in, const int* in_sizes, int n_in,
                              void* d_out, int out_size, void* d_ws, size_t ws_size,
                              hipStream_t stream) {
    (void)in_sizes; (void)n_in; (void)out_size; (void)ws_size;
    const float* hidden = (const float*)d_in[0];
    // d_in[1] = cu_seqlens_q: fixed [0,1024,2048,3072,4096] -> 1024-token segments
    const float* Wq = (const float*)d_in[2];
    const float* Wk = (const float*)d_in[3];
    const float* Wv = (const float*)d_in[4];
    const float* Wo = (const float*)d_in[5];
    float* out = (float*)d_out;

    const int nHid = TOKENS * HDIM;
    const int nWq  = HDIM * HDIM;
    const int nWk  = KVDIM * HDIM;

    char* ws = (char*)d_ws;
    _Float16* hid_h = (_Float16*)ws;  ws += (size_t)nHid * 2;
    _Float16* Wq_h  = (_Float16*)ws;  ws += (size_t)nWq  * 2;
    _Float16* Wk_h  = (_Float16*)ws;  ws += (size_t)nWk  * 2;
    _Float16* Wv_h  = (_Float16*)ws;  ws += (size_t)nWk  * 2;
    _Float16* Wo_h  = (_Float16*)ws;  ws += (size_t)nWq  * 2;
    _Float16* Qh    = (_Float16*)ws;  ws += (size_t)TOKENS * HDIM  * 2;
    _Float16* Kh    = (_Float16*)ws;  ws += (size_t)TOKENS * KVDIM * 2;
    _Float16* Vh    = (_Float16*)ws;  ws += (size_t)TOKENS * KVDIM * 2;
    _Float16* Ah    = (_Float16*)ws;  ws += (size_t)TOKENS * HDIM  * 2;

    f32_to_f16_kernel<<<(nHid + 255) / 256, 256, 0, stream>>>(hidden, hid_h, nHid);
    f32_to_f16_kernel<<<(nWq  + 255) / 256, 256, 0, stream>>>(Wq, Wq_h, nWq);
    f32_to_f16_kernel<<<(nWk  + 255) / 256, 256, 0, stream>>>(Wk, Wk_h, nWk);
    f32_to_f16_kernel<<<(nWk  + 255) / 256, 256, 0, stream>>>(Wv, Wv_h, nWk);
    f32_to_f16_kernel<<<(nWq  + 255) / 256, 256, 0, stream>>>(Wo, Wo_h, nWq);

    wmma_gemm_nt<_Float16><<<dim3(HDIM / 128, TOKENS / 128), 256, 0, stream>>>(
        hid_h, Wq_h, Qh, TOKENS, HDIM, HDIM);
    wmma_gemm_nt<_Float16><<<dim3(KVDIM / 128, TOKENS / 128), 256, 0, stream>>>(
        hid_h, Wk_h, Kh, TOKENS, KVDIM, HDIM);
    wmma_gemm_nt<_Float16><<<dim3(KVDIM / 128, TOKENS / 128), 256, 0, stream>>>(
        hid_h, Wv_h, Vh, TOKENS, KVDIM, HDIM);

    // 4 segments * 64 q-tiles * 32 heads = 8192 waves, 2 waves/block
    attn_kernel<<<8192 / 2, 64, 0, stream>>>(Qh, Kh, Vh, Ah);

    wmma_gemm_nt<float><<<dim3(HDIM / 128, TOKENS / 128), 256, 0, stream>>>(
        Ah, Wo_h, out, TOKENS, HDIM, HDIM);
}
